// Conv2D_30296699306350
// MI455X (gfx1250) — compile-verified, matrix operand order checked by reading, not run on hardware
//
#include <hip/hip_runtime.h>

typedef __attribute__((ext_vector_type(16))) __bf16 v16bf;
typedef __attribute__((ext_vector_type(8)))  float  v8f;
typedef __attribute__((ext_vector_type(4)))  int    v4i;

#define IN_C   128
#define OUT_C  256
#define NIMG   32
#define Hh     56
#define Ww     56
#define HW     (Hh * Ww)              // 3136
#define KTOT   (IN_C * 9)             // 1152
#define PW     58                     // padded width
#define PHW    (PW * PW)              // 3364

#define MTILE  128                    // out-channels per block
#define NTILE  64                     // spatial: 8x8 output patch
#define KSTEP  32
#define LDA    40                     // padded LDS stride (bf16 elems)
#define LDB    40

// ---- workspace layout (bytes) ----
#define WSW_ELEMS  (OUT_C * KTOT)               // 294912 weight elems
#define WS_WHI_OFF 0
#define WS_WLO_OFF (WSW_ELEMS * 2)              // 589824
#define WS_I32_OFF (WSW_ELEMS * 4)              // 1179648
#define PAD_ELEMS  (NIMG * IN_C * PHW)          // 13778944 padded input elems

// ---- CDNA5 async global->LDS path (guarded) ----
#if defined(__has_builtin)
#  if __has_builtin(__builtin_amdgcn_global_load_async_to_lds_b128)
#    define HAS_ASYNC 1
#  endif
#  if __has_builtin(__builtin_amdgcn_s_wait_asynccnt)
#    define HAS_WAITASYNC 1
#  endif
#endif
#ifndef HAS_ASYNC
#  define HAS_ASYNC 0
#endif
#ifndef HAS_WAITASYNC
#  define HAS_WAITASYNC 0
#endif

__device__ __forceinline__ unsigned short f32_to_bf16(float f) {
    unsigned u = __float_as_uint(f);
    unsigned r = u + 0x7FFFu + ((u >> 16) & 1u);    // round-to-nearest-even
    return (unsigned short)(r >> 16);
}
__device__ __forceinline__ float bf16_to_f32(unsigned short h) {
    return __uint_as_float(((unsigned)h) << 16);
}

__device__ __forceinline__ void copy16_g2l(unsigned short* ldst,
                                           const unsigned short* gsrc) {
#if HAS_ASYNC
    // signature (probe-learned): (v4i AS1*, v4i AS3*, imm offset, imm cpol)
    __builtin_amdgcn_global_load_async_to_lds_b128(
        (__attribute__((address_space(1))) v4i*)(v4i*)gsrc,
        (__attribute__((address_space(3))) v4i*)(v4i*)ldst, 0, 0);
#else
    *(uint4*)ldst = *(const uint4*)gsrc;
#endif
}
__device__ __forceinline__ void wait_async_all() {
#if HAS_ASYNC
#  if HAS_WAITASYNC
    __builtin_amdgcn_s_wait_asynccnt(0);
#  else
    asm volatile("s_wait_asynccnt 0" ::: "memory");
#  endif
#endif
}

union FragU { uint4 q[2]; v16bf v; };

// ---------------- prepass 1: weights fp32 -> hi/lo bf16 planes --------------
__global__ __launch_bounds__(256)
void convert_weights_kernel(const float* __restrict__ w,
                            unsigned short* __restrict__ whi,
                            unsigned short* __restrict__ wlo)
{
    int idx = (blockIdx.x * 256 + threadIdx.x) * 4;     // 288 blocks, exact
    float4 f = *(const float4*)(w + idx);
    float v[4] = { f.x, f.y, f.z, f.w };
    unsigned short h[4], l[4];
#pragma unroll
    for (int i = 0; i < 4; ++i) {
        h[i] = f32_to_bf16(v[i]);
        l[i] = f32_to_bf16(v[i] - bf16_to_f32(h[i]));
    }
    *(uint2*)(whi + idx) = make_uint2((unsigned)h[0] | ((unsigned)h[1] << 16),
                                      (unsigned)h[2] | ((unsigned)h[3] << 16));
    *(uint2*)(wlo + idx) = make_uint2((unsigned)l[0] | ((unsigned)l[1] << 16),
                                      (unsigned)l[2] | ((unsigned)l[3] << 16));
}

// ------- prepass 2: input fp32 NCHW -> zero-padded 58x58, hi|lo packed ------
__global__ __launch_bounds__(256)
void convert_input_kernel(const float* __restrict__ in,
                          unsigned int* __restrict__ ipk)
{
    unsigned idx = blockIdx.x * 256 + threadIdx.x;      // 53824 blocks, exact
    unsigned x    = idx % PW;
    unsigned rest = idx / PW;
    unsigned y    = rest % PW;
    unsigned nc   = rest / PW;
    float v = 0.0f;
    if (x >= 1u && x <= (unsigned)Ww && y >= 1u && y <= (unsigned)Hh)
        v = in[(size_t)nc * HW + (y - 1) * Ww + (x - 1)];
    unsigned short h = f32_to_bf16(v);
    unsigned short l = f32_to_bf16(v - bf16_to_f32(h));
    ipk[idx] = (unsigned)h | ((unsigned)l << 16);
}

// ---------------- main: implicit-GEMM conv, split-bf16 WMMA -----------------
__global__ __launch_bounds__(256)
void conv2d_wmma_kernel(const unsigned short* __restrict__ whi,
                        const unsigned short* __restrict__ wlo,
                        const unsigned int* __restrict__ ipk,
                        float* __restrict__ out)
{
    __shared__ __align__(16) unsigned short smAhi[MTILE * LDA];
    __shared__ __align__(16) unsigned short smAlo[MTILE * LDA];
    __shared__ __align__(16) unsigned short smBhi[NTILE * LDB];
    __shared__ __align__(16) unsigned short smBlo[NTILE * LDB];
    __shared__ __align__(16) int smOff[KTOT];           // im2col addr offsets

    const int t       = threadIdx.x;
    const int tx      = blockIdx.x % 7;
    const int ty      = blockIdx.x / 7;
    const int ow_base = tx * 8;
    const int oh_base = ty * 8;
    const int oc_base = blockIdx.y * MTILE;
    const int nimg    = blockIdx.z;

    // one-time im2col offset table: off(k) = c*3364 + kh*58 + kw
    for (int k = t; k < KTOT; k += 256) {
        int c  = k / 9;
        int r9 = k - 9 * c;
        int kh = r9 / 3;
        int kw = r9 - 3 * kh;
        smOff[k] = c * PHW + kh * PW + kw;
    }
    __syncthreads();

    const int wave = t >> 5, lane = t & 31;
    const int mloc = lane & 15;
    const int lo_k = (lane < 16) ? 0 : 8;    // ISA 16-bit fragment K layout
    const int wm   = wave >> 1;              // 0..3 : 32-row M slab
    const int wn   = wave & 1;               // 0..1 : 32-col N slab

    // B fill: thread owns one spatial pos p, 8 contiguous k
    const int b_p  = t >> 2;                 // 0..63
    const int b_kq = (t & 3) << 3;           // 0,8,16,24
    const int b_py = b_p >> 3, b_px = b_p & 7;
    const unsigned int* ibase =
        ipk + (size_t)(nimg * IN_C) * PHW
            + (unsigned)((oh_base + b_py) * PW + (ow_base + b_px));

    // A fill: 512 16-byte chunks/plane, 2 per thread per plane
    const int a_rowA = t >> 2;               // rows 0..63
    const int a_rowB = 64 + (t >> 2);        // rows 64..127
    const int a_kc   = (t & 3) << 3;         // 0,8,16,24
    const unsigned short* wHiBase = whi + (size_t)oc_base * KTOT;
    const unsigned short* wLoBase = wlo + (size_t)oc_base * KTOT;

    v8f acc00 = {}, acc01 = {}, acc10 = {}, acc11 = {};

    const unsigned short* aHi0 = smAhi + (wm * 32 + mloc) * LDA + lo_k;
    const unsigned short* aLo0 = smAlo + (wm * 32 + mloc) * LDA + lo_k;
    const unsigned short* aHi1 = aHi0 + 16 * LDA;
    const unsigned short* aLo1 = aLo0 + 16 * LDA;
    const unsigned short* bHi0 = smBhi + (wn * 32 + mloc) * LDB + lo_k;
    const unsigned short* bLo0 = smBlo + (wn * 32 + mloc) * LDB + lo_k;
    const unsigned short* bHi1 = bHi0 + 16 * LDB;
    const unsigned short* bLo1 = bLo0 + 16 * LDB;

    for (int k0 = 0; k0 < KTOT; k0 += KSTEP) {
        // ---- stage A (pre-converted weights): async global->LDS ----
        copy16_g2l(smAhi + a_rowA * LDA + a_kc, wHiBase + a_rowA * KTOT + k0 + a_kc);
        copy16_g2l(smAhi + a_rowB * LDA + a_kc, wHiBase + a_rowB * KTOT + k0 + a_kc);
        copy16_g2l(smAlo + a_rowA * LDA + a_kc, wLoBase + a_rowA * KTOT + k0 + a_kc);
        copy16_g2l(smAlo + a_rowB * LDA + a_kc, wLoBase + a_rowB * KTOT + k0 + a_kc);

        // ---- stage B^T (packed padded input): gather, split hi/lo ----
        {
            int4 o0 = *(const int4*)&smOff[k0 + b_kq];
            int4 o1 = *(const int4*)&smOff[k0 + b_kq + 4];
            unsigned u0 = ibase[o0.x], u1 = ibase[o0.y];
            unsigned u2 = ibase[o0.z], u3 = ibase[o0.w];
            unsigned u4 = ibase[o1.x], u5 = ibase[o1.y];
            unsigned u6 = ibase[o1.z], u7 = ibase[o1.w];
            *(uint4*)(smBhi + b_p * LDB + b_kq) =
                make_uint4((u0 & 0xffffu) | (u1 << 16),
                           (u2 & 0xffffu) | (u3 << 16),
                           (u4 & 0xffffu) | (u5 << 16),
                           (u6 & 0xffffu) | (u7 << 16));
            *(uint4*)(smBlo + b_p * LDB + b_kq) =
                make_uint4((u0 >> 16) | (u1 & 0xffff0000u),
                           (u2 >> 16) | (u3 & 0xffff0000u),
                           (u4 >> 16) | (u5 & 0xffff0000u),
                           (u6 >> 16) | (u7 & 0xffff0000u));
        }
        wait_async_all();
        __syncthreads();

        // ---- fragments (b128 pairs per ISA 16-bit layout) ----
        FragU ah0, al0, ah1, al1, bh0, bl0, bh1, bl1;
        ah0.q[0] = *(const uint4*)(aHi0);  ah0.q[1] = *(const uint4*)(aHi0 + 16);
        al0.q[0] = *(const uint4*)(aLo0);  al0.q[1] = *(const uint4*)(aLo0 + 16);
        ah1.q[0] = *(const uint4*)(aHi1);  ah1.q[1] = *(const uint4*)(aHi1 + 16);
        al1.q[0] = *(const uint4*)(aLo1);  al1.q[1] = *(const uint4*)(aLo1 + 16);
        bh0.q[0] = *(const uint4*)(bHi0);  bh0.q[1] = *(const uint4*)(bHi0 + 16);
        bl0.q[0] = *(const uint4*)(bLo0);  bl0.q[1] = *(const uint4*)(bLo0 + 16);
        bh1.q[0] = *(const uint4*)(bHi1);  bh1.q[1] = *(const uint4*)(bHi1 + 16);
        bl1.q[0] = *(const uint4*)(bLo1);  bl1.q[1] = *(const uint4*)(bLo1 + 16);

        // ---- split-bf16: hi*hi + hi*lo + lo*hi per C tile (12 WMMA) ----
        acc00 = __builtin_amdgcn_wmma_f32_16x16x32_bf16(false, ah0.v, false, bh0.v, (short)0, acc00, false, false);
        acc00 = __builtin_amdgcn_wmma_f32_16x16x32_bf16(false, ah0.v, false, bl0.v, (short)0, acc00, false, false);
        acc00 = __builtin_amdgcn_wmma_f32_16x16x32_bf16(false, al0.v, false, bh0.v, (short)0, acc00, false, false);

        acc01 = __builtin_amdgcn_wmma_f32_16x16x32_bf16(false, ah0.v, false, bh1.v, (short)0, acc01, false, false);
        acc01 = __builtin_amdgcn_wmma_f32_16x16x32_bf16(false, ah0.v, false, bl1.v, (short)0, acc01, false, false);
        acc01 = __builtin_amdgcn_wmma_f32_16x16x32_bf16(false, al0.v, false, bh1.v, (short)0, acc01, false, false);

        acc10 = __builtin_amdgcn_wmma_f32_16x16x32_bf16(false, ah1.v, false, bh0.v, (short)0, acc10, false, false);
        acc10 = __builtin_amdgcn_wmma_f32_16x16x32_bf16(false, ah1.v, false, bl0.v, (short)0, acc10, false, false);
        acc10 = __builtin_amdgcn_wmma_f32_16x16x32_bf16(false, al1.v, false, bh0.v, (short)0, acc10, false, false);

        acc11 = __builtin_amdgcn_wmma_f32_16x16x32_bf16(false, ah1.v, false, bh1.v, (short)0, acc11, false, false);
        acc11 = __builtin_amdgcn_wmma_f32_16x16x32_bf16(false, ah1.v, false, bl1.v, (short)0, acc11, false, false);
        acc11 = __builtin_amdgcn_wmma_f32_16x16x32_bf16(false, al1.v, false, bh1.v, (short)0, acc11, false, false);

        __syncthreads();
    }

    // ---- write back: ISA 16x16 f32 C layout (VGPR r: M=r / r+8; N=lane%16) --
    const int oc0 = oc_base + wm * 32 + ((lane < 16) ? 0 : 8);
    const int pl0 = wn * 32 + mloc;
    const int pl1 = pl0 + 16;
    const size_t sp0 = (size_t)(oh_base + (pl0 >> 3)) * Ww + (ow_base + (pl0 & 7));
    const size_t sp1 = (size_t)(oh_base + (pl1 >> 3)) * Ww + (ow_base + (pl1 & 7));
#pragma unroll
    for (int r = 0; r < 8; ++r) {
        size_t o = (size_t)(nimg * OUT_C + oc0 + r) * HW;
        out[o + sp0]                    = acc00[r];
        out[o + sp1]                    = acc01[r];
        out[o + (size_t)16 * HW + sp0]  = acc10[r];
        out[o + (size_t)16 * HW + sp1]  = acc11[r];
    }
}

extern "C" void kernel_launch(void* const* d_in, const int* in_sizes, int n_in,
                              void* d_out, int out_size, void* d_ws, size_t ws_size,
                              hipStream_t stream) {
    (void)in_sizes; (void)n_in; (void)out_size; (void)ws_size;
    const float* in  = (const float*)d_in[0];
    const float* wgt = (const float*)d_in[1];
    float* out = (float*)d_out;

    unsigned short* whi = (unsigned short*)((char*)d_ws + WS_WHI_OFF);
    unsigned short* wlo = (unsigned short*)((char*)d_ws + WS_WLO_OFF);
    unsigned int*   ipk = (unsigned int*)((char*)d_ws + WS_I32_OFF);

    // prepasses (same stream -> ordered before main kernel)
    hipLaunchKernelGGL(convert_weights_kernel, dim3(WSW_ELEMS / 1024), dim3(256),
                       0, stream, wgt, whi, wlo);
    hipLaunchKernelGGL(convert_input_kernel, dim3(PAD_ELEMS / 256), dim3(256),
                       0, stream, in, ipk);

    hipLaunchKernelGGL(conv2d_wmma_kernel, dim3(49, 2, NIMG), dim3(256),
                       0, stream, whi, wlo, ipk, out);
}
